// LSTM_70592082477746
// MI455X (gfx1250) — matrix operand order repaired and checked by reading, not compiled, not run on hardware
//
#include <hip/hip_runtime.h>

// MI455X (gfx1250) LSTM: bf16 WMMA GEMMs + f32 accumulate, wave32.
// d_out layout: [SEQ*B*HID] output, then [B*HID] h_final. No workspace used.
// Phase 1 writes Xg into d_out; phase 2 (1 WG, serial) overwrites Xg[t] with
// h_t (double-buffered async Xg staging in LDS); phase 3 overwrites h with y.

typedef __attribute__((ext_vector_type(16))) __bf16 v16bf;
typedef __attribute__((ext_vector_type(8)))  float  v8f;
typedef __attribute__((ext_vector_type(4)))  int    v4i;

#define SEQ   2048
#define NB    32
#define NH    256
#define ROWS  (SEQ * NB)        // 65536 flattened (t, b) rows
#define KP    264               // padded K stride (bf16 elems) for LDS weights
#define HP    264               // padded row stride (f32 elems) for LDS h

#define HAS_ASYNC_LDS  __has_builtin(__builtin_amdgcn_global_load_async_to_lds_b128)
#define HAS_WAIT_ASYNC __has_builtin(__builtin_amdgcn_s_wait_asynccnt)

#if HAS_WAIT_ASYNC
#define WAIT_ASYNC(n) __builtin_amdgcn_s_wait_asynccnt(n)
#elif HAS_ASYNC_LDS
#define WAIT_ASYNC(n) asm volatile("s_wait_asynccnt " #n ::: "memory")
#else
#define WAIT_ASYNC(n)
#endif

// ---- fast scalar math on the gfx1250 TRANS pipe (guarded, with fallbacks) --
__device__ __forceinline__ float fast_rcp(float x) {
#if __has_builtin(__builtin_amdgcn_rcpf)
  return __builtin_amdgcn_rcpf(x);
#else
  return 1.0f / x;
#endif
}
__device__ __forceinline__ float fast_exp(float x) {
#if __has_builtin(__builtin_amdgcn_exp2f)
  return __builtin_amdgcn_exp2f(x * 1.4426950408889634f);
#else
  return __expf(x);
#endif
}
__device__ __forceinline__ float fast_tanh(float x) {
#if __has_builtin(__builtin_amdgcn_tanhf)
  return __builtin_amdgcn_tanhf(x);            // V_TANH_F32
#else
  float e = fast_exp(2.0f * x);
  return 1.0f - 2.0f * fast_rcp(e + 1.0f);
#endif
}
__device__ __forceinline__ float fast_sigmoid(float x) {
  return fast_rcp(1.0f + fast_exp(-x));
}

__device__ __forceinline__ v8f wmma_bf16(v16bf a, v16bf b, v8f c) {
  return __builtin_amdgcn_wmma_f32_16x16x32_bf16(
      false, a, false, b, (short)0, c, false, false);
}

// Stage a 256x256 f32 weight (row-major [K][N]) into LDS as bf16,
// transposed to [N][KP] so B-fragments read two runs of 8 contiguous K per lane.
__device__ __forceinline__ void stage_weights(const float* __restrict__ W,
                                              __bf16* Wt, int tid) {
  for (int e = tid; e < NH * NH; e += 256) {
    int k = e >> 8;
    int n = e & 255;
    Wt[n * KP + k] = (__bf16)W[e];
  }
}

// A-fragment (16x32 bf16) from global f32, row-major stride NH.
__device__ __forceinline__ v16bf load_afrag_g(const float* __restrict__ base,
                                              int kblock, int lane) {
  int r  = lane & 15;
  int kb = kblock + ((lane >> 4) << 3);
  const float* p = base + r * NH + kb;
  v16bf a;
#pragma unroll
  for (int i = 0; i < 8; ++i) { a[i] = (__bf16)p[i]; a[i + 8] = (__bf16)p[i + 16]; }
  return a;
}

// Same A-fragment sourced from f32 h in LDS (row stride HP).
__device__ __forceinline__ v16bf load_afrag_l(const float* h, int m0,
                                              int kblock, int lane) {
  int r  = m0 + (lane & 15);
  int kb = kblock + ((lane >> 4) << 3);
  const float* p = h + r * HP + kb;
  v16bf a;
#pragma unroll
  for (int i = 0; i < 8; ++i) { a[i] = (__bf16)p[i]; a[i + 8] = (__bf16)p[i + 16]; }
  return a;
}

// B-fragment (32x16 bf16) from LDS weights [N][KP]: two runs of 8 K per lane (k = 8h+i, 16+8h+i).
__device__ __forceinline__ v16bf load_bfrag(const __bf16* Wt, int n0,
                                            int kblock, int lane) {
  const __bf16* p = Wt + (n0 + (lane & 15)) * KP + kblock + ((lane >> 4) << 3);
  v16bf b;
#pragma unroll
  for (int i = 0; i < 8; ++i) { b[i] = p[i]; b[i + 8] = p[i + 16]; }
  return b;
}

#if HAS_ASYNC_LDS
// Async-copy a 32x256 f32 tile (32 KB) global -> LDS; 8 b128 ops per thread,
// i.e. ASYNCcnt += 8 per wave.
__device__ __forceinline__ void issue_tile_copy(const float* g, float* l, int tid) {
#pragma unroll
  for (int i = 0; i < 8; ++i) {
    int chunk = (tid + (i << 8)) << 2;        // 16B chunk -> 4 floats
    __builtin_amdgcn_global_load_async_to_lds_b128(
        (__attribute__((address_space(1))) v4i*)(g + chunk),
        (__attribute__((address_space(3))) v4i*)(l + chunk), 0, 0);
  }
}
#endif

// ---------------- Phase 1: Xg = X @ W_x + b_gate (into d_out) --------------
__global__ void __launch_bounds__(256, 1)
lstm_gate_precompute(const float* __restrict__ x,
                     const float* __restrict__ Wg,
                     const float* __restrict__ bg,
                     float* __restrict__ Xg) {
  extern __shared__ char smem[];
  __bf16* Wt    = (__bf16*)smem;                             // 256*KP bf16
  float*  biasl = (float*)(smem + NH * KP * sizeof(__bf16)); // 256 f32
  int tid = threadIdx.x;
  stage_weights(Wg, Wt, tid);                 // rows 0..255 of W_gate = W_x
  biasl[tid] = bg[tid];
  __syncthreads();

  int wave = tid >> 5, lane = tid & 31;
  size_t r0 = (size_t)blockIdx.x * 128 + (size_t)wave * 16;
  const float* abase = x + r0 * NH;
  int col16 = lane & 15;
  int rsub  = (lane >> 4) << 3;

  v16bf afr[8];
#pragma unroll
  for (int f = 0; f < 8; ++f) afr[f] = load_afrag_g(abase, 32 * f, lane);

#pragma unroll 1
  for (int nt = 0; nt < 16; ++nt) {
    int n0 = nt << 4;
    float bv = biasl[n0 + col16];             // folds into dscnt wait stream
    v16bf bfr[8];
#pragma unroll
    for (int f = 0; f < 8; ++f) bfr[f] = load_bfrag(Wt, n0, 32 * f, lane);
    v8f acc;
#pragma unroll
    for (int v = 0; v < 8; ++v) acc[v] = bv;
#pragma unroll
    for (int f = 0; f < 8; ++f) acc = wmma_bf16(afr[f], bfr[f], acc);
    int col = n0 + col16;
#pragma unroll
    for (int v = 0; v < 8; ++v) Xg[(r0 + rsub + v) * NH + col] = acc[v];
  }
}

// ------- Phase 2: serial recurrence, 1 WG; overwrites Xg[t] with h_t -------
__global__ void __launch_bounds__(256, 1)
lstm_recurrent(const float* __restrict__ Wg,
               float* __restrict__ dout) {
  extern __shared__ char smem[];
  __bf16* Wt  = (__bf16*)smem;                               // 256*KP bf16
  float*  hsh = (float*)(smem + NH * KP * sizeof(__bf16));   // 32 x HP f32
  float*  xb  = (float*)(smem + NH * KP * sizeof(__bf16)
                              + NB * HP * sizeof(float));    // 2 x 32*256 f32
  int tid = threadIdx.x;

#if HAS_ASYNC_LDS
  issue_tile_copy(dout, xb, tid);            // prime: tile 0 -> xb parity 0
#endif
  stage_weights(Wg + NH * NH, Wt, tid);      // rows 256..511 of W_gate = W_h
  for (int e = tid; e < NB * HP; e += 256) hsh[e] = 0.0f;

  int wave = tid >> 5, lane = tid & 31;
  int mt  = wave & 1;
  int ntg = wave >> 1;
  int m0  = mt << 4;
  int col16 = lane & 15;
  int rsub  = (lane >> 4) << 3;

  float cst[4][8];
#pragma unroll
  for (int q = 0; q < 4; ++q)
#pragma unroll
    for (int v = 0; v < 8; ++v) cst[q][v] = 0.0f;

  __syncthreads();

#pragma unroll 1
  for (int t = 0; t < SEQ; ++t) {
    float* xg = dout + (size_t)t * NB * NH;

    v16bf afr[8];                            // h_{t-1} -> bf16 A fragments
#pragma unroll
    for (int f = 0; f < 8; ++f) afr[f] = load_afrag_l(hsh, m0, 32 * f, lane);

#if HAS_ASYNC_LDS
    if (t + 1 < SEQ) {
      issue_tile_copy(xg + NB * NH, xb + ((t + 1) & 1) * (NB * NH), tid);
      WAIT_ASYNC(8);                         // tile t resident in LDS
    } else {
      WAIT_ASYNC(0);
    }
    const float* xsrc = xb + (t & 1) * (NB * NH);
#else
    __builtin_prefetch(xg + NB * NH + tid * 32, 0, 1);   // global_prefetch_b8
    const float* xsrc = xg;
#endif
    __syncthreads();                         // h reads done + xb[t] visible

#pragma unroll
    for (int q = 0; q < 4; ++q) {
      int n0 = ((ntg << 2) + q) << 4;
      v16bf bfr[8];
#pragma unroll
      for (int f = 0; f < 8; ++f) bfr[f] = load_bfrag(Wt, n0, 32 * f, lane);
      v8f acc;
#pragma unroll
      for (int v = 0; v < 8; ++v) acc[v] = 0.0f;
#pragma unroll
      for (int f = 0; f < 8; ++f) acc = wmma_bf16(afr[f], bfr[f], acc);

      int col = n0 + col16;
#pragma unroll
      for (int v = 0; v < 8; ++v) {
        int row = m0 + rsub + v;
        float g    = acc[v] + xsrc[row * NH + col];   // Xg already has b_gate
        float gate = fast_sigmoid(g);
        float cv   = cst[q][v] * gate + fast_tanh(g) * gate;
        cst[q][v]  = cv;
        float hv   = fast_tanh(cv) * gate;
        hsh[row * HP + col] = hv;            // h for next step's A
        xg [row * NH + col] = hv;            // h_t for phase 3 (in d_out)
      }
    }
    __syncthreads();                         // h writes visible
  }

  for (int e = tid; e < NB * NH; e += 256)
    dout[(size_t)ROWS * NH + e] = hsh[(e >> 8) * HP + (e & 255)];
}

// -------- Phase 3: y = h @ W_out + b_out, in place over d_out --------------
__global__ void __launch_bounds__(256, 1)
lstm_out_proj(float* __restrict__ Y,
              const float* __restrict__ Wo,
              const float* __restrict__ bo) {
  extern __shared__ char smem[];
  __bf16* Wt    = (__bf16*)smem;
  float*  biasl = (float*)(smem + NH * KP * sizeof(__bf16));
  int tid = threadIdx.x;
  stage_weights(Wo, Wt, tid);
  biasl[tid] = bo[tid];
  __syncthreads();

  int wave = tid >> 5, lane = tid & 31;
  size_t r0 = (size_t)blockIdx.x * 128 + (size_t)wave * 16;
  int col16 = lane & 15;
  int rsub  = (lane >> 4) << 3;

  v16bf afr[8];                              // load ALL h rows before storing
#pragma unroll
  for (int f = 0; f < 8; ++f) afr[f] = load_afrag_g(Y + r0 * NH, 32 * f, lane);

#pragma unroll 1
  for (int nt = 0; nt < 16; ++nt) {
    int n0 = nt << 4;
    float bv = biasl[n0 + col16];
    v16bf bfr[8];
#pragma unroll
    for (int f = 0; f < 8; ++f) bfr[f] = load_bfrag(Wt, n0, 32 * f, lane);
    v8f acc;
#pragma unroll
    for (int v = 0; v < 8; ++v) acc[v] = bv;
#pragma unroll
    for (int f = 0; f < 8; ++f) acc = wmma_bf16(afr[f], bfr[f], acc);
    int col = n0 + col16;
#pragma unroll
    for (int v = 0; v < 8; ++v) Y[(r0 + rsub + v) * NH + col] = acc[v];
  }
}

extern "C" void kernel_launch(void* const* d_in, const int* in_sizes, int n_in,
                              void* d_out, int out_size, void* d_ws, size_t ws_size,
                              hipStream_t stream) {
  (void)in_sizes; (void)n_in; (void)out_size; (void)d_ws; (void)ws_size;
  const float* x  = (const float*)d_in[0];
  const float* Wg = (const float*)d_in[1];
  const float* bg = (const float*)d_in[2];
  const float* Wo = (const float*)d_in[3];
  const float* bo = (const float*)d_in[4];
  float* out = (float*)d_out;

  size_t ldsW = (size_t)NH * KP * sizeof(__bf16)
              + (size_t)NH * sizeof(float);                   // Wt + bias
  size_t ldsR = (size_t)NH * KP * sizeof(__bf16)
              + (size_t)NB * HP * sizeof(float)
              + (size_t)2 * NB * NH * sizeof(float);          // Wt + h + xb

  lstm_gate_precompute<<<ROWS / 128, 256, ldsW, stream>>>(x, Wg, bg, out);
  lstm_recurrent      <<<1,          256, ldsR, stream>>>(Wg, out);
  lstm_out_proj       <<<ROWS / 128, 256, ldsW, stream>>>(out, Wo, bo);
}
